// MultiHeadAttention_59992103190747
// MI455X (gfx1250) — matrix operand order repaired and checked by reading, not compiled, not run on hardware
//
#include <hip/hip_runtime.h>
#include <hip/hip_bf16.h>

// ---------------------------------------------------------------------------
// MultiHeadAttention for MI455X (gfx1250):
//  - all GEMMs on v_wmma_f32_16x16x32_f16
//  - attention K/V tiles staged by the Tensor Data Mover (tensor_load_to_lds,
//    TENSORcnt), double-buffered in LDS, HW row padding = LDS pitch
//  - softmax cross-lane reductions via ds_swizzle_b32 (immediate xor pattern)
// ---------------------------------------------------------------------------

typedef __attribute__((ext_vector_type(16))) _Float16 v16h;
typedef __attribute__((ext_vector_type(8)))  _Float16 v8h;
typedef __attribute__((ext_vector_type(8)))  float    v8f;
typedef __attribute__((ext_vector_type(4)))  float    v4f;
typedef __attribute__((ext_vector_type(4)))  uint32_t u32x4;
typedef __attribute__((ext_vector_type(8)))  uint32_t u32x8;

static constexpr int Bb  = 4;
static constexpr int Ss  = 2048;
static constexpr int Dd  = 1024;
static constexpr int Hh  = 16;
static constexpr int DKk = 64;

static constexpr int TM  = 64;       // GEMM tile M
static constexpr int TN  = 64;       // GEMM tile N
static constexpr int TK  = 32;       // GEMM tile K
static constexpr int LDT = TK + 8;   // 40 halves = 80 B rows (16B multiple)
static constexpr int LDH = DKk + 16; // 80 halves = 160 B rows (16B multiple)

__device__ __forceinline__ v8f zero_v8f() {
  v8f r;
#pragma unroll
  for (int i = 0; i < 8; ++i) r[i] = 0.0f;
  return r;
}

__device__ __forceinline__ v16h concat_v16h(v8h lo, v8h hi) {
  v16h r;
#pragma unroll
  for (int i = 0; i < 8; ++i) { r[i] = lo[i]; r[8 + i] = hi[i]; }
  return r;
}

// A fragment (16x32, M x K), row-major source with leading dim `ld` (halves).
__device__ __forceinline__ v16h load_frag_a(const _Float16* base, int ld) {
  const int l  = threadIdx.x & 31;
  const int m  = l & 15;
  const int kh = (l < 16) ? 0 : 8;
  const _Float16* p = base + m * ld + kh;
  v8h lo = *(const v8h*)(p);
  v8h hi = *(const v8h*)(p + 16);
  return concat_v16h(lo, hi);
}

// B fragment (32x16, K x N) from B^T storage (row = N, col = K, lead dim ld).
__device__ __forceinline__ v16h load_frag_b(const _Float16* bt, int ld) {
  const int l  = threadIdx.x & 31;
  const int n  = l & 15;
  const int kb = (l < 16) ? 0 : 8;
  const _Float16* p = bt + n * ld + kb;
  v8h lo = *(const v8h*)(p);
  v8h hi = *(const v8h*)(p + 16);
  return concat_v16h(lo, hi);
}

__device__ __forceinline__ v8f wmma16(v16h a, v16h b, v8f c) {
  return __builtin_amdgcn_wmma_f32_16x16x32_f16(false, a, false, b,
                                                (short)0, c, false, false);
}

// Cross-lane xor exchange via ds_swizzle_b32, group-of-32 mode:
// offset = {1'b0, and_mask=0x1f, or_mask=0, xor_mask=MASK}.
// (lane & 0x1f) ^ MASK, MASK <= 8 stays within each 16-lane row group.
template <int MASK>
__device__ __forceinline__ float swz_xor(float x) {
  int r = __builtin_amdgcn_ds_swizzle(__float_as_int(x), (0x1f << 10) | MASK);
  return __int_as_float(r);
}
__device__ __forceinline__ float red_max16(float x) {
  x = fmaxf(x, swz_xor<1>(x));
  x = fmaxf(x, swz_xor<2>(x));
  x = fmaxf(x, swz_xor<4>(x));
  x = fmaxf(x, swz_xor<8>(x));
  return x;
}
__device__ __forceinline__ float red_sum16(float x) {
  x += swz_xor<1>(x);
  x += swz_xor<2>(x);
  x += swz_xor<4>(x);
  x += swz_xor<8>(x);
  return x;
}

// ---------------------------------------------------------------------------
// Tensor Data Mover: 2-D tile load (global -> LDS), f16 elements.
// D# per CDNA5 ISA. Groups 2/3 omitted (2-D tensor).
// Row padding: tile rows are 128 B; pad_interval=4 (128 B), pad_amount=7
// (8 dwords = 32 B) -> LDS row pitch 160 B == LDH halves.
// ---------------------------------------------------------------------------
__device__ __forceinline__ void tdm_load_2d_f16(uint32_t lds_off,
                                                const void* gaddr,
                                                uint32_t dim0,      // elems/row
                                                uint32_t dim1,      // rows
                                                uint32_t stride0,   // elems
                                                uint32_t tile0,     // elems
                                                uint32_t tile1) {   // rows
  uint64_t ga = (uint64_t)(uintptr_t)gaddr;
  u32x4 g0;
  g0[0] = 1u;                                           // count=1, user mode
  g0[1] = lds_off;                                      // lds_addr
  g0[2] = (uint32_t)ga;                                 // global_addr[31:0]
  g0[3] = (uint32_t)((ga >> 32) & 0x01FFFFFFu) | (2u << 30);  // [56:32]|type=2
  u32x8 g1;
  g1[0] = (1u << 16) |                                  // data_size = 2 bytes
          (1u << 20) |                                  // pad_enable
          (4u << 22) |                                  // pad_interval: 128 B
          (7u << 25);                                   // pad_amount: 32 B
  g1[1] = (dim0 & 0xFFFFu) << 16;                       // tensor_dim0[15:0]
  g1[2] = (dim0 >> 16) | ((dim1 & 0xFFFFu) << 16);      // dim0 hi | dim1 lo
  g1[3] = (dim1 >> 16) | (tile0 << 16);                 // dim1 hi | tile_dim0
  g1[4] = tile1;                                        // tile_dim1 (dim2=0)
  g1[5] = stride0;                                      // dim0_stride[31:0]
  g1[6] = 0u;                                           // stride hi, dim1_stride
  g1[7] = 0u;
  asm volatile("tensor_load_to_lds %0, %1" :: "s"(g0), "s"(g1) : "memory");
}
__device__ __forceinline__ void wait_tensor0() {
  asm volatile("s_wait_tensorcnt 0x0" ::: "memory");
}

// ---------------------------------------------------------------------------
// Tiled GEMM:  out = A[M,K] @ W[K,N] + bias
// MODE 0: f16 out, head layout   [B,H,S,DK]      (Q, K projections)
// MODE 1: f16 out, transposed    [B,H,DK,S]      (V projection)
// MODE 2: f32 out, plain         [M,N]           (output projection)
// ---------------------------------------------------------------------------
template <typename AT, int MODE>
__global__ void __launch_bounds__(128, 1)
gemm64_kernel(const AT* __restrict__ A,
              const float* __restrict__ W,
              const float* __restrict__ bias,
              void* __restrict__ out,
              int M, int N, int Kd) {
  __shared__ __align__(16) _Float16 As[TM][LDT];
  __shared__ __align__(16) _Float16 Bt[TN][LDT];

  const int m0  = blockIdx.x * TM;
  const int n0  = blockIdx.y * TN;
  const int tid = threadIdx.x;
  const int w   = tid >> 5;
  const int l   = tid & 31;

  // Staging maps (coalesced 16-consecutive-element reads per thread).
  const int ar  = tid >> 1;            // A row 0..63
  const int ac0 = (tid & 1) * 16;      // A col 0/16
  const int wk  = tid >> 2;            // W row (k) 0..31
  const int wn0 = (tid & 3) * 16;      // W col (n) 0/16/32/48

  v8f acc[4];
#pragma unroll
  for (int t = 0; t < 4; ++t) acc[t] = zero_v8f();

  for (int kb = 0; kb < Kd; kb += TK) {
    __syncthreads();
    const AT*    asrc = A + (size_t)(m0 + ar) * Kd + kb + ac0;
    const float* wsrc = W + (size_t)(kb + wk) * N + n0 + wn0;
    if (kb + TK < Kd) {  // prefetch next tiles into cache
      __builtin_prefetch((const void*)(asrc + TK), 0, 1);
      __builtin_prefetch((const void*)(wsrc + (size_t)TK * N), 0, 1);
    }
    // A tile: 16 elems/thread as two 8-wide chunks -> single b128 LDS stores.
#pragma unroll
    for (int j = 0; j < 2; ++j) {
      v8h hval;
      if constexpr (sizeof(AT) == 2) {
        hval = *(const v8h*)((const _Float16*)asrc + j * 8);
      } else {
        v4f x0 = *(const v4f*)((const float*)asrc + j * 8);
        v4f x1 = *(const v4f*)((const float*)asrc + j * 8 + 4);
#pragma unroll
        for (int e = 0; e < 4; ++e) {
          hval[e]     = (_Float16)x0[e];
          hval[4 + e] = (_Float16)x1[e];
        }
      }
      *(v8h*)&As[ar][ac0 + j * 8] = hval;
    }
    // W tile: coalesced row reads, transposed scatter into Bt.
#pragma unroll
    for (int i = 0; i < 16; ++i)
      Bt[wn0 + i][wk] = (_Float16)wsrc[i];
    __syncthreads();

    v16h a = load_frag_a(&As[w * 16][0], LDT);
#pragma unroll
    for (int t = 0; t < 4; ++t) {
      v16h b = load_frag_b(&Bt[t * 16][0], LDT);
      acc[t] = wmma16(a, b, acc[t]);
    }
  }

  // Epilogue: C layout — lane holds col N=l&15, rows v + (l<16 ? 0 : 8).
  const int ro = (l < 16) ? 0 : 8;
  const int nl = l & 15;
#pragma unroll
  for (int t = 0; t < 4; ++t) {
#pragma unroll
    for (int v = 0; v < 8; ++v) {
      int gr = m0 + w * 16 + v + ro;
      int gc = n0 + t * 16 + nl;
      float val = acc[t][v] + bias[gc];
      if (MODE == 2) {
        ((float*)out)[(size_t)gr * N + gc] = val;
      } else {
        int b = gr / Ss, s = gr % Ss;
        int h = gc / DKk, dk = gc % DKk;
        size_t off = (MODE == 0)
            ? (((size_t)(b * Hh + h) * Ss + s) * DKk + dk)
            : (((size_t)(b * Hh + h) * DKk + dk) * Ss + s);
        ((_Float16*)out)[off] = (_Float16)val;
      }
    }
  }
}

// ---------------------------------------------------------------------------
// Flash attention: grid (S/64, B*H), block 128 (4 waves x 16 query rows).
// Q,K: [B,H,S,DK] f16.  Vt: [B,H,DK,S] f16.  O: [B,S,D] f16.
// K/V tiles double-buffered via TDM (wave 0 issues; EXEC ignored by TDM).
// ---------------------------------------------------------------------------
__global__ void __launch_bounds__(128, 1)
attention64_kernel(const _Float16* __restrict__ Q,
                   const _Float16* __restrict__ Kt,
                   const _Float16* __restrict__ Vt,
                   _Float16* __restrict__ O) {
  __shared__ __align__(16) _Float16 Ks[2][64][LDH];    // rows s_k, cols dk
  __shared__ __align__(16) _Float16 Vs[2][64][LDH];    // rows dk,  cols s_k
  __shared__ __align__(16) _Float16 Pw[4][16][LDH];    // per-wave P tile

  const int bh  = blockIdx.y;
  const int b   = bh / Hh;
  const int h   = bh % Hh;
  const int qb  = blockIdx.x * 64;
  const int tid = threadIdx.x;
  const int w   = tid >> 5;
  const int l   = tid & 31;

  const _Float16* kbase = Kt + (size_t)bh * Ss * DKk;
  const _Float16* vbase = Vt + (size_t)bh * DKk * Ss;

  // Q fragments for this wave's 16 rows, pre-scaled by 1/sqrt(DK).
  const _Float16* qp = Q + ((size_t)bh * Ss + qb + w * 16) * DKk;
  v16h qf[2];
  qf[0] = load_frag_a(qp, DKk);
  qf[1] = load_frag_a(qp + 32, DKk);
#pragma unroll
  for (int d = 0; d < 2; ++d)
#pragma unroll
    for (int i = 0; i < 16; ++i)
      qf[d][i] = (_Float16)((float)qf[d][i] * 0.125f);

  v8f acc[4];
#pragma unroll
  for (int t = 0; t < 4; ++t) acc[t] = zero_v8f();
  float mi[8], li[8];
#pragma unroll
  for (int v = 0; v < 8; ++v) { mi[v] = -1e30f; li[v] = 0.0f; }

  // One TDM descriptor per K tile and per V tile (wave 0 only).
  auto stage = [&](int buf, int kb) {
    if (w == 0) {
      tdm_load_2d_f16((uint32_t)(uintptr_t)&Ks[buf][0][0],
                      kbase + (size_t)kb * DKk,
                      /*dim0=*/DKk, /*dim1=*/Ss, /*stride0=*/DKk,
                      /*tile0=*/64, /*tile1=*/64);
      tdm_load_2d_f16((uint32_t)(uintptr_t)&Vs[buf][0][0],
                      vbase + kb,
                      /*dim0=*/Ss, /*dim1=*/DKk, /*stride0=*/Ss,
                      /*tile0=*/64, /*tile1=*/64);
    }
  };

  stage(0, 0);
  if (w == 0) wait_tensor0();
  __syncthreads();

  int buf = 0;
  for (int kb = 0; kb < Ss; kb += 64) {
    if (kb + 64 < Ss) stage(buf ^ 1, kb + 64);   // TDM prefetch next tile

    // Scores: S = (Q * scale) @ K^T  — 16x64 per wave, 8 WMMAs.
    v8f sc[4];
#pragma unroll
    for (int t = 0; t < 4; ++t) {
      sc[t] = zero_v8f();
#pragma unroll
      for (int d = 0; d < 2; ++d)
        sc[t] = wmma16(qf[d], load_frag_b(&Ks[buf][t * 16][d * 32], LDH), sc[t]);
    }

    // Online softmax per row (lanes 0..15 / 16..31 share row groups).
#pragma unroll
    for (int v = 0; v < 8; ++v) {
      float rmax = fmaxf(fmaxf(sc[0][v], sc[1][v]), fmaxf(sc[2][v], sc[3][v]));
      rmax = red_max16(rmax);
      float mnew  = fmaxf(mi[v], rmax);
      float alpha = __expf(mi[v] - mnew);
      mi[v] = mnew;
      li[v] *= alpha;
      float rsum = 0.0f;
#pragma unroll
      for (int t = 0; t < 4; ++t) {
        float p = __expf(sc[t][v] - mnew);
        sc[t][v] = p;
        rsum += p;
        acc[t][v] *= alpha;
      }
      li[v] += red_sum16(rsum);
    }

    // C-layout -> A-fragment layout via per-wave LDS round trip.
    {
      const int ro = (l < 16) ? 0 : 8;
      const int nl = l & 15;
#pragma unroll
      for (int t = 0; t < 4; ++t)
#pragma unroll
        for (int v = 0; v < 8; ++v)
          Pw[w][v + ro][t * 16 + nl] = (_Float16)sc[t][v];
    }
    asm volatile("s_wait_dscnt 0x0" ::: "memory");

    v16h pf0 = load_frag_a(&Pw[w][0][0], LDH);
    v16h pf1 = load_frag_a(&Pw[w][0][32], LDH);
#pragma unroll
    for (int t = 0; t < 4; ++t) {
      acc[t] = wmma16(pf0, load_frag_b(&Vs[buf][t * 16][0], LDH), acc[t]);
      acc[t] = wmma16(pf1, load_frag_b(&Vs[buf][t * 16][32], LDH), acc[t]);
    }

    if (w == 0) wait_tensor0();   // next tile's DMA complete (issuing wave)
    __syncthreads();              // make it visible to all waves
    buf ^= 1;
  }

  // Normalize and store to [B,S,D] f16.
  const int ro = (l < 16) ? 0 : 8;
  const int nl = l & 15;
#pragma unroll
  for (int v = 0; v < 8; ++v) {
    float inv = 1.0f / li[v];
    int s = qb + w * 16 + v + ro;
#pragma unroll
    for (int t = 0; t < 4; ++t) {
      int col = h * DKk + t * 16 + nl;
      O[((size_t)b * Ss + s) * Dd + col] = (_Float16)(acc[t][v] * inv);
    }
  }
}

// ---------------------------------------------------------------------------
extern "C" void kernel_launch(void* const* d_in, const int* in_sizes, int n_in,
                              void* d_out, int out_size, void* d_ws, size_t ws_size,
                              hipStream_t stream) {
  const float* x  = (const float*)d_in[0];
  const float* Wq = (const float*)d_in[1];
  const float* bq = (const float*)d_in[2];
  const float* Wk = (const float*)d_in[3];
  const float* bk = (const float*)d_in[4];
  const float* Wv = (const float*)d_in[5];
  const float* bv = (const float*)d_in[6];
  const float* Wo = (const float*)d_in[7];
  const float* bo = (const float*)d_in[8];
  float* out = (float*)d_out;

  const size_t NE = (size_t)Bb * Ss * Dd;  // 8,388,608 elements
  _Float16* Qb = (_Float16*)d_ws;
  _Float16* Kb = Qb + NE;
  _Float16* Vb = Kb + NE;
  _Float16* Ab = Vb + NE;

  const int M = Bb * Ss;   // 8192
  const int N = Dd;        // 1024
  const int K = Dd;        // 1024

  dim3 gg(M / TM, N / TN);                 // 128 x 16
  dim3 ga(Ss / 64, Bb * Hh);               // 32 x 64

  gemm64_kernel<float, 0><<<gg, 128, 0, stream>>>(x, Wq, bq, (void*)Qb, M, N, K);
  gemm64_kernel<float, 0><<<gg, 128, 0, stream>>>(x, Wk, bk, (void*)Kb, M, N, K);
  gemm64_kernel<float, 1><<<gg, 128, 0, stream>>>(x, Wv, bv, (void*)Vb, M, N, K);
  attention64_kernel<<<ga, 128, 0, stream>>>(Qb, Kb, Vb, Ab);
  gemm64_kernel<_Float16, 2><<<gg, 128, 0, stream>>>(Ab, Wo, bo, (void*)out, M, N, K);
}
